// GraphVampNet_73624329388106
// MI455X (gfx1250) — matrix operand order repaired and checked by reading, not compiled, not run on hardware
//
#include <hip/hip_runtime.h>
#include <hip/hip_bf16.h>

typedef __bf16 v16bf __attribute__((ext_vector_type(16)));
typedef __bf16 v8bf  __attribute__((ext_vector_type(8)));
typedef float  v8f   __attribute__((ext_vector_type(8)));

#define Bn 32
#define Nn 512
#define Kn 16
#define Hh 64
#define Ll 4
#define Cc 6

// ---------------------------------------------------------------------------
// WMMA operand loaders (wave32, v_wmma_f32_16x16x32_bf16)
// A 16x32 bf16 layout (ISA 7.12.2): lane<16 -> row=lane, K {0..7,16..23};
//                                   lane>=16 -> row=lane-16, K {8..15,24..31}
// ---------------------------------------------------------------------------
__device__ __forceinline__ v16bf lds_loadA_bf16(const __bf16* base, int ld,
                                                int row0, int k0, int lane) {
  int half = lane >> 4, r = lane & 15;
  const __bf16* p = base + (row0 + r) * ld + k0 + half * 8;
  v8bf lo = *(const v8bf*)(p);
  v8bf hi = *(const v8bf*)(p + 16);
  v16bf a;
#pragma unroll
  for (int i = 0; i < 8; ++i) { a[i] = lo[i]; a[i + 8] = hi[i]; }
  return a;
}

__device__ __forceinline__ v16bf lds_loadA_f32(const float* base, int ld,
                                               int row0, int k0, int lane) {
  int half = lane >> 4, r = lane & 15;
  const float* p = base + (row0 + r) * ld + k0 + half * 8;
  v16bf a;
#pragma unroll
  for (int i = 0; i < 8; ++i) { a[i] = (__bf16)p[i]; a[i + 8] = (__bf16)p[i + 16]; }
  return a;
}

// B 32x16 bf16, pre-packed so lane n reads its 16 bf16 contiguously (32B).
__device__ __forceinline__ v16bf loadB_packed(const __bf16* tile, int lane) {
  return ((const v16bf*)tile)[lane];
}

// silu via fast hardware reciprocal (v_rcp_f32) instead of IEEE div chains
__device__ __forceinline__ float silu(float v) {
  return v * __builtin_amdgcn_rcpf(1.0f + __expf(-v));
}

__device__ __forceinline__ unsigned pack2bf(float a, float b) {
  union { __bf16 h[2]; unsigned u; } p;
  p.h[0] = (__bf16)a; p.h[1] = (__bf16)b;
  return p.u;
}

// ---------------------------------------------------------------------------
// Weight repack: src [L][Kdim x 64] row-major f32 -> packed bf16 WMMA-B tiles.
// Tile index ((l*KT + kt)*4 + nt), 512 bf16 per tile; element (lane,i) holds
// W[k = kt*32 + (i>=8)*16 + (lane>=16)*8 + (i&7)][col = nt*16 + (lane&15)], zero-padded.
// ---------------------------------------------------------------------------
__global__ void pack_w(const float* __restrict__ src, __bf16* __restrict__ dst,
                       int Ksrc, int KT, int layerStride) {
  int idx = blockIdx.x * 256 + threadIdx.x;
  int total = Ll * KT * 4 * 512;
  if (idx >= total) return;
  int pos  = idx & 511;
  int tile = idx >> 9;
  int nt = tile & 3;
  int kt = (tile >> 2) % KT;
  int l  = tile / (KT * 4);
  int lane = pos >> 4, i = pos & 15;
  int col = (lane & 15) + nt * 16;
  int k   = kt * 32 + ((i >> 3) << 4) + ((lane >> 4) << 3) + (i & 7);
  float v = (k < Ksrc) ? src[l * layerStride + k * Hh + col] : 0.0f;
  dst[idx] = (__bf16)v;
}

// h0[b*N+i][c] = (emb[i] @ W_in)[c] + b_in[c], broadcast over batch
__global__ void init_h(const float* __restrict__ emb, const float* __restrict__ Win,
                       const float* __restrict__ bin, float* __restrict__ h) {
  int idx = blockIdx.x * 256 + threadIdx.x;
  if (idx >= Nn * Hh) return;
  int i = idx >> 6, c = idx & 63;
  float s = bin[c];
#pragma unroll 8
  for (int k = 0; k < Hh; ++k) s += emb[i * Hh + k] * Win[k * Hh + c];
  for (int b = 0; b < Bn; ++b) h[(b * Nn + i) * Hh + c] = s;
}

// x[bn][d] = data[bn*(3+N)+d], stored with stride 4
__global__ void init_x(const float* __restrict__ data, float* __restrict__ x) {
  int idx = blockIdx.x * 256 + threadIdx.x;
  if (idx >= Bn * Nn * 3) return;
  int bn = idx / 3, d = idx - bn * 3;
  x[bn * 4 + d] = data[bn * (3 + Nn) + d];
}

// ---------------------------------------------------------------------------
// One E_GCL layer. Block = 16 nodes of one batch (256 edges), 256 threads.
// ---------------------------------------------------------------------------
__global__ __launch_bounds__(256) void egcl_layer(
    int l,
    const float* __restrict__ h_in, const float* __restrict__ x_in,
    float* __restrict__ h_out, float* __restrict__ x_out,
    const __bf16* __restrict__ pw_ew1, const __bf16* __restrict__ pw_ew2,
    const __bf16* __restrict__ pw_nw1, const __bf16* __restrict__ pw_nw2,
    const __bf16* __restrict__ pw_cw1,
    const float* __restrict__ edge_b1, const float* __restrict__ edge_b2,
    const float* __restrict__ node_b1, const float* __restrict__ node_b2,
    const float* __restrict__ coord_b1, const float* __restrict__ coord_w2) {
  __shared__ __attribute__((aligned(16))) float  smemA[64 * 160 / 2]; // 20KB
  __shared__ __attribute__((aligned(16))) __bf16 hiddenL[64 * 64];    // 8KB
  __shared__ __attribute__((aligned(16))) float  m32[64 * 64];        // 16KB
  __shared__ __attribute__((aligned(16))) __bf16 hbL[32 * 64];        // 4KB
  __shared__ __attribute__((aligned(16))) float  hownL[16 * 64];      // 4KB
  __shared__ float xsL[32 * 4];
  __shared__ float aggL[16 * 64];
  __shared__ float tsumL[16 * 4];
  __shared__ float transL[64 * 4];

  const int tid  = threadIdx.x;
  const int lane = tid & 31;
  const int wv   = tid >> 5;
  const int b    = blockIdx.x >> 5;          // 32 node-tiles per batch
  const int i0   = (blockIdx.x & 31) << 4;
  const int g0   = b * Nn;

  __bf16* m_inL = (__bf16*)smemA;            // 64 x 160 bf16 (per chunk)
  float*  ph32  = smemA;                     // 64 x 64 f32 (aliased after GEMM1)
  __bf16* n_inL = (__bf16*)smemA;            // 16 x 128 bf16 (phase 2)
  unsigned* m_inU = (unsigned*)smemA;        // dword view: row stride 80
  const unsigned* hbU = (const unsigned*)hbL; // dword view: row stride 32

  const float*  eb1 = edge_b1  + l * Hh;
  const float*  eb2 = edge_b2  + l * Hh;
  const float*  nb1 = node_b1  + l * Hh;
  const float*  nb2 = node_b2  + l * Hh;
  const float*  cb1 = coord_b1 + l * Hh;
  const float*  cw2 = coord_w2 + l * Hh;
  const __bf16* ew1 = pw_ew1 + l * 5 * 4 * 512;
  const __bf16* ew2 = pw_ew2 + l * 2 * 4 * 512;
  const __bf16* nw1 = pw_nw1 + l * 4 * 4 * 512;
  const __bf16* nw2 = pw_nw2 + l * 2 * 4 * 512;
  const __bf16* cw1 = pw_cw1 + l * 2 * 4 * 512;

  // ---- Phase 0: stage h rows i0..i0+31 (bf16), own rows (f32), x rows ----
  for (int idx = tid; idx < 32 * Hh; idx += 256) {
    int r = idx >> 6, c = idx & 63;
    float v = h_in[(g0 + ((i0 + r) & (Nn - 1))) * Hh + c];
    hbL[idx] = (__bf16)v;
    if (r < 16) hownL[idx] = v;
  }
  if (tid < 128) {
    int r = tid >> 2, d = tid & 3;
    xsL[tid] = (d < 3) ? x_in[(g0 + ((i0 + r) & (Nn - 1))) * 4 + d] : 0.0f;
  }
  for (int idx = tid; idx < 16 * Hh; idx += 256) aggL[idx] = 0.0f;
  if (tid < 64) tsumL[tid] = 0.0f;
  __syncthreads();

  // ---- Edge path: 4 chunks x 64 edges ----
  for (int ch = 0; ch < 4; ++ch) {
    // build m_in = [h_row | h_col | radial,1 | 0pad] (64 x 160 bf16),
    // three branch-free dword passes (row stride 80 dwords)
    for (int idx = tid; idx < 64 * 32; idx += 256) {      // cols 0..63 (h_row)
      int row = idx >> 5, u = idx & 31;
      int a = ch * 4 + (row >> 4);
      m_inU[row * 80 + u] = hbU[a * 32 + u];
    }
    for (int idx = tid; idx < 64 * 32; idx += 256) {      // cols 64..127 (h_col)
      int row = idx >> 5, u = idx & 31;
      int a = ch * 4 + (row >> 4);
      int cf = a + 1 + (row & 15);
      m_inU[row * 80 + 32 + u] = hbU[cf * 32 + u];
    }
    for (int idx = tid; idx < 64 * 16; idx += 256) {      // cols 128..159 tail
      int row = idx >> 4, u = idx & 15;
      unsigned v = 0u;
      if (u == 0) {
        int a = ch * 4 + (row >> 4);
        int cf = a + 1 + (row & 15);
        float dx = xsL[a * 4 + 0] - xsL[cf * 4 + 0];
        float dy = xsL[a * 4 + 1] - xsL[cf * 4 + 1];
        float dz = xsL[a * 4 + 2] - xsL[cf * 4 + 2];
        v = pack2bf(dx * dx + dy * dy + dz * dz, 1.0f);
      }
      m_inU[row * 80 + 64 + u] = v;
    }
    __syncthreads();

    // GEMM1: hidden = silu(m_in[64x160] @ w1 + b1) -> bf16, 16 tiles / 8 waves
    for (int t = wv * 2; t < wv * 2 + 2; ++t) {
      int mt = t >> 2, nt = t & 3;
      v8f acc = {};
      for (int kt = 0; kt < 5; ++kt) {
        v16bf av = lds_loadA_bf16(m_inL, 160, mt * 16, kt * 32, lane);
        v16bf bv = loadB_packed(ew1 + (kt * 4 + nt) * 512, lane);
        acc = __builtin_amdgcn_wmma_f32_16x16x32_bf16(false, av, false, bv,
                                                      (short)0, acc, false, false);
      }
      int col = nt * 16 + (lane & 15);
      int rb  = mt * 16 + ((lane >> 4) << 3);
      float bias = eb1[col];
#pragma unroll
      for (int r = 0; r < 8; ++r)
        hiddenL[(rb + r) * 64 + col] = (__bf16)silu(acc[r] + bias);
    }
    __syncthreads();

    // GEMM2: m = silu(hidden @ w2 + b2) -> f32 in m32
    for (int t = wv * 2; t < wv * 2 + 2; ++t) {
      int mt = t >> 2, nt = t & 3;
      v8f acc = {};
      for (int kt = 0; kt < 2; ++kt) {
        v16bf av = lds_loadA_bf16(hiddenL, 64, mt * 16, kt * 32, lane);
        v16bf bv = loadB_packed(ew2 + (kt * 4 + nt) * 512, lane);
        acc = __builtin_amdgcn_wmma_f32_16x16x32_bf16(false, av, false, bv,
                                                      (short)0, acc, false, false);
      }
      int col = nt * 16 + (lane & 15);
      int rb  = mt * 16 + ((lane >> 4) << 3);
      float bias = eb2[col];
#pragma unroll
      for (int r = 0; r < 8; ++r)
        m32[(rb + r) * 64 + col] = silu(acc[r] + bias);
    }
    __syncthreads();

    // agg[node] += sum over the node's 16 edges (exact, fp32)
    {
      int q = tid >> 6, c = tid & 63;
      float s = 0.0f;
#pragma unroll
      for (int r = 0; r < 16; ++r) s += m32[(q * 16 + r) * 64 + c];
      aggL[(ch * 4 + q) * 64 + c] += s;
    }
    __syncthreads();

    // GEMM3: ph = silu(m @ cw1 + cb1) -> f32 (aliased onto smemA)
    for (int t = wv * 2; t < wv * 2 + 2; ++t) {
      int mt = t >> 2, nt = t & 3;
      v8f acc = {};
      for (int kt = 0; kt < 2; ++kt) {
        v16bf av = lds_loadA_f32(m32, 64, mt * 16, kt * 32, lane);
        v16bf bv = loadB_packed(cw1 + (kt * 4 + nt) * 512, lane);
        acc = __builtin_amdgcn_wmma_f32_16x16x32_bf16(false, av, false, bv,
                                                      (short)0, acc, false, false);
      }
      int col = nt * 16 + (lane & 15);
      int rb  = mt * 16 + ((lane >> 4) << 3);
      float bias = cb1[col];
#pragma unroll
      for (int r = 0; r < 8; ++r)
        ph32[(rb + r) * 64 + col] = silu(acc[r] + bias);
    }
    __syncthreads();

    // phi = ph @ cw2 (H->1), trans = diff * phi
    if (tid < 64) {
      int e = tid, q = e >> 4, k = e & 15;
      int a = ch * 4 + q, cf = a + 1 + k;
      float phi = 0.0f;
#pragma unroll 8
      for (int c = 0; c < Hh; ++c) phi += ph32[e * 64 + c] * cw2[c];
      transL[e * 4 + 0] = (xsL[a * 4 + 0] - xsL[cf * 4 + 0]) * phi;
      transL[e * 4 + 1] = (xsL[a * 4 + 1] - xsL[cf * 4 + 1]) * phi;
      transL[e * 4 + 2] = (xsL[a * 4 + 2] - xsL[cf * 4 + 2]) * phi;
    }
    __syncthreads();
    if (tid < 12) {
      int q = tid / 3, d = tid - q * 3;
      float s = 0.0f;
#pragma unroll
      for (int r = 0; r < 16; ++r) s += transL[(q * 16 + r) * 4 + d];
      tsumL[(ch * 4 + q) * 4 + d] += s;
    }
    __syncthreads();
  }

  // ---- Phase 2: node update h += mlp2([h, agg]); x += tsum/16 ----
  for (int idx = tid; idx < 16 * 128; idx += 256) {
    int row = idx >> 7, col = idx & 127;
    float v = (col < 64) ? hownL[row * 64 + col] : aggL[row * 64 + (col - 64)];
    n_inL[idx] = (__bf16)v;
  }
  __syncthreads();
  if (wv < 4) {   // wave-uniform; EXEC all-ones inside
    int nt = wv;
    v8f acc = {};
    for (int kt = 0; kt < 4; ++kt) {
      v16bf av = lds_loadA_bf16(n_inL, 128, 0, kt * 32, lane);
      v16bf bv = loadB_packed(nw1 + (kt * 4 + nt) * 512, lane);
      acc = __builtin_amdgcn_wmma_f32_16x16x32_bf16(false, av, false, bv,
                                                    (short)0, acc, false, false);
    }
    int col = nt * 16 + (lane & 15);
    int rb  = (lane >> 4) << 3;
    float bias = nb1[col];
#pragma unroll
    for (int r = 0; r < 8; ++r)
      hiddenL[(rb + r) * 64 + col] = (__bf16)silu(acc[r] + bias);
  }
  __syncthreads();
  if (wv < 4) {
    int nt = wv;
    v8f acc = {};
    for (int kt = 0; kt < 2; ++kt) {
      v16bf av = lds_loadA_bf16(hiddenL, 64, 0, kt * 32, lane);
      v16bf bv = loadB_packed(nw2 + (kt * 4 + nt) * 512, lane);
      acc = __builtin_amdgcn_wmma_f32_16x16x32_bf16(false, av, false, bv,
                                                    (short)0, acc, false, false);
    }
    int col = nt * 16 + (lane & 15);
    int rb  = (lane >> 4) << 3;
    float bias = nb2[col];
#pragma unroll
    for (int r = 0; r < 8; ++r) {
      int row = rb + r;
      h_out[(g0 + i0 + row) * Hh + col] = acc[r] + bias + hownL[row * 64 + col];
    }
  }
  if (tid < 48) {
    int r = tid / 3, d = tid - r * 3;
    x_out[(g0 + i0 + r) * 4 + d] = xsL[r * 4 + d] + tsumL[r * 4 + d] * (1.0f / 16.0f);
  }
}

// ---------------------------------------------------------------------------
// Readout: mean over N, @W_out+b_out, @W_cls+b_cls, softmax -> [B,6]
// ---------------------------------------------------------------------------
__global__ void readout(const float* __restrict__ h,
                        const float* __restrict__ Wout, const float* __restrict__ bout,
                        const float* __restrict__ Wcls, const float* __restrict__ bcls,
                        float* __restrict__ out) {
  __shared__ float mh[Hh], prot[Hh], lg[Cc];
  int b = blockIdx.x, t = threadIdx.x;
  float s = 0.0f;
  for (int n = 0; n < Nn; ++n) s += h[(b * Nn + n) * Hh + t];
  mh[t] = s * (1.0f / Nn);
  __syncthreads();
  float p = bout[t];
#pragma unroll 8
  for (int k = 0; k < Hh; ++k) p += mh[k] * Wout[k * Hh + t];
  prot[t] = p;
  __syncthreads();
  if (t < Cc) {
    float v = bcls[t];
    for (int c = 0; c < Hh; ++c) v += prot[c] * Wcls[c * Cc + t];
    lg[t] = v;
  }
  __syncthreads();
  if (t == 0) {
    float mx = lg[0];
    for (int j = 1; j < Cc; ++j) mx = fmaxf(mx, lg[j]);
    float e[Cc], se = 0.0f;
    for (int j = 0; j < Cc; ++j) { e[j] = __expf(lg[j] - mx); se += e[j]; }
    float rs = __builtin_amdgcn_rcpf(se);
    for (int j = 0; j < Cc; ++j) out[b * Cc + j] = e[j] * rs;
  }
}

// ---------------------------------------------------------------------------
extern "C" void kernel_launch(void* const* d_in, const int* in_sizes, int n_in,
                              void* d_out, int out_size, void* d_ws, size_t ws_size,
                              hipStream_t stream) {
  const float* data     = (const float*)d_in[0];
  const float* emb      = (const float*)d_in[1];
  const float* W_in     = (const float*)d_in[2];
  const float* b_in     = (const float*)d_in[3];
  const float* edge_w1  = (const float*)d_in[4];
  const float* edge_b1  = (const float*)d_in[5];
  const float* edge_w2  = (const float*)d_in[6];
  const float* edge_b2  = (const float*)d_in[7];
  const float* node_w1  = (const float*)d_in[8];
  const float* node_b1  = (const float*)d_in[9];
  const float* node_w2  = (const float*)d_in[10];
  const float* node_b2  = (const float*)d_in[11];
  const float* coord_w1 = (const float*)d_in[12];
  const float* coord_b1 = (const float*)d_in[13];
  const float* coord_w2 = (const float*)d_in[14];
  const float* W_out    = (const float*)d_in[15];
  const float* b_out    = (const float*)d_in[16];
  const float* W_cls    = (const float*)d_in[17];
  const float* b_cls    = (const float*)d_in[18];
  (void)in_sizes; (void)n_in; (void)out_size; (void)ws_size;

  char* ws = (char*)d_ws;
  float* hA = (float*)(ws + 0);                  // 4 MB
  float* hB = (float*)(ws + 4194304);            // 4 MB
  float* xA = (float*)(ws + 8388608);            // 256 KB (stride-4)
  float* xB = (float*)(ws + 8650752);            // 256 KB
  __bf16* pw     = (__bf16*)(ws + 8912896);
  __bf16* pw_ew1 = pw;                           // L*5*4*512 = 40960 elems
  __bf16* pw_ew2 = pw_ew1 + Ll * 5 * 4 * 512;    // 16384
  __bf16* pw_nw1 = pw_ew2 + Ll * 2 * 4 * 512;    // 32768
  __bf16* pw_nw2 = pw_nw1 + Ll * 4 * 4 * 512;    // 16384
  __bf16* pw_cw1 = pw_nw2 + Ll * 2 * 4 * 512;    // 16384

  pack_w<<<(Ll * 5 * 4 * 512 + 255) / 256, 256, 0, stream>>>(edge_w1,  pw_ew1, 130, 5, 130 * Hh);
  pack_w<<<(Ll * 2 * 4 * 512 + 255) / 256, 256, 0, stream>>>(edge_w2,  pw_ew2, 64,  2, 64 * Hh);
  pack_w<<<(Ll * 4 * 4 * 512 + 255) / 256, 256, 0, stream>>>(node_w1,  pw_nw1, 128, 4, 128 * Hh);
  pack_w<<<(Ll * 2 * 4 * 512 + 255) / 256, 256, 0, stream>>>(node_w2,  pw_nw2, 64,  2, 64 * Hh);
  pack_w<<<(Ll * 2 * 4 * 512 + 255) / 256, 256, 0, stream>>>(coord_w1, pw_cw1, 64,  2, 64 * Hh);
  init_h<<<(Nn * Hh + 255) / 256, 256, 0, stream>>>(emb, W_in, b_in, hA);
  init_x<<<(Bn * Nn * 3 + 255) / 256, 256, 0, stream>>>(data, xA);

  float* hi = hA; float* ho = hB; float* xi = xA; float* xo = xB;
  for (int l = 0; l < Ll; ++l) {
    egcl_layer<<<Bn * (Nn / 16), 256, 0, stream>>>(
        l, hi, xi, ho, xo, pw_ew1, pw_ew2, pw_nw1, pw_nw2, pw_cw1,
        edge_b1, edge_b2, node_b1, node_b2, coord_b1, coord_w2);
    float* t;
    t = hi; hi = ho; ho = t;
    t = xi; xi = xo; xo = t;
  }
  readout<<<Bn, Hh, 0, stream>>>(hi, W_out, b_out, W_cls, b_cls, (float*)d_out);
}
